// HighLevelGuidedFrequencyFusion_61057255080370
// MI455X (gfx1250) — matrix operand order repaired and checked by reading, not compiled, not run on hardware
//
#include <hip/hip_runtime.h>
#include <hip/hip_bf16.h>

typedef __attribute__((ext_vector_type(16))) _Float16 v16h;
typedef __attribute__((ext_vector_type(8)))  float    v8f;

#define Bb 4
#define Cc 64
#define Hh 256
#define Ww 256
#define TOK 1024
#define SFEAT 2114
#define NTOK 4096
#define KSEL 1024
#define NHh 4
#define S1 ((long)Bb*Cc*Hh*Ww)   // 16,777,216 floats

// ---------------- WMMA fragment helpers (wave32, 16x16x32 f16) ----------------
__device__ inline v16h frag_a16(const _Float16* sA, int lane) {
  v16h a; unsigned* au = (unsigned*)&a;
  int m = lane & 15, hk = (lane >> 4) * 16;
#pragma unroll
  for (int j = 0; j < 8; j++) {
    int kk = hk + j * 2;
    au[j] = *(const unsigned*)(sA + m * 32 + kk);
  }
  return a;
}
__device__ inline v16h frag_b16(const _Float16* sB, int lane) {
  v16h b; unsigned* bu = (unsigned*)&b;
  int n = lane & 15, kb = (lane >> 4) * 16;
#pragma unroll
  for (int j = 0; j < 8; j++) bu[j] = *(const unsigned*)(sB + n * 32 + kb + 2 * j);
  return b;
}

// ---------------- Generic batched WMMA GEMM ----------------
// C = alpha * A @ op(B) + beta * C.  A row-major [M,K]; B row-major [K,N] (transB=0)
// or B[n*ldb+k] (transB=1). M,N mult of 32; K mult of 32. Batch off: (bz/div)*s1+(bz%div)*s2.
struct GemmArgs {
  const float* A; const float* B; float* C;
  int M, N, K, lda, ldb, ldc, transB;
  float alpha, beta;
  int div;
  long a1, a2, b1, b2, c1, c2;
};

__global__ __launch_bounds__(128) void gemm_kernel(GemmArgs g) {
  int tid = threadIdx.x, wid = tid >> 5, lane = tid & 31;
  int bz = blockIdx.z;
  const float* A = g.A + (long)(bz / g.div) * g.a1 + (long)(bz % g.div) * g.a2;
  const float* B = g.B + (long)(bz / g.div) * g.b1 + (long)(bz % g.div) * g.b2;
  float*       C = g.C + (long)(bz / g.div) * g.c1 + (long)(bz % g.div) * g.c2;
  int mb = blockIdx.y * 32, nb = blockIdx.x * 32;
  __shared__ __align__(16) _Float16 sA[2][1024];   // [tile][chunk*512 + row*32 + k]
  __shared__ __align__(16) _Float16 sB[2][1024];   // [tile][chunk*512 + n*32 + k]
  int wm = wid >> 1, wn = wid & 1;
  v8f acc = {};
  int k0 = 0;
  // ---- main loop: 64-deep K slabs, 2 WMMA per barrier pair ----
  for (; k0 + 64 <= g.K; k0 += 64) {
    if (wid < 2) {
      int row = lane & 15, kq = (lane >> 4) * 16;
      const float* p = A + (long)(mb + wid * 16 + row) * g.lda + k0 + kq;
#pragma unroll
      for (int c2 = 0; c2 < 2; c2++) {
        const float* pc = p + c2 * 32;
        _Float16* d = &sA[wid][c2 * 512 + row * 32 + kq];
#pragma unroll
        for (int i = 0; i < 4; i++) {
          float4 v = *(const float4*)(pc + i * 4);
          d[i * 4 + 0] = (_Float16)v.x; d[i * 4 + 1] = (_Float16)v.y;
          d[i * 4 + 2] = (_Float16)v.z; d[i * 4 + 3] = (_Float16)v.w;
        }
      }
      __builtin_prefetch(p + 64, 0, 3);   // next K slab -> global_prefetch_b8
    } else {
      int t = wid - 2;
      if (!g.transB) {
        const float* p = B + (long)(k0 + lane) * g.ldb + nb + t * 16;
#pragma unroll
        for (int c2 = 0; c2 < 2; c2++) {
          const float* pc = p + (long)c2 * 32 * g.ldb;
#pragma unroll
          for (int i = 0; i < 4; i++) {
            float4 v = *(const float4*)(pc + i * 4);
            sB[t][c2 * 512 + (i * 4 + 0) * 32 + lane] = (_Float16)v.x;
            sB[t][c2 * 512 + (i * 4 + 1) * 32 + lane] = (_Float16)v.y;
            sB[t][c2 * 512 + (i * 4 + 2) * 32 + lane] = (_Float16)v.z;
            sB[t][c2 * 512 + (i * 4 + 3) * 32 + lane] = (_Float16)v.w;
          }
        }
        __builtin_prefetch(p + (long)64 * g.ldb, 0, 3);
      } else {
        int row = lane & 15, kq = (lane >> 4) * 16;
        const float* p = B + (long)(nb + t * 16 + row) * g.ldb + k0 + kq;
#pragma unroll
        for (int c2 = 0; c2 < 2; c2++) {
          const float* pc = p + c2 * 32;
          _Float16* d = &sB[t][c2 * 512 + row * 32 + kq];
#pragma unroll
          for (int i = 0; i < 4; i++) {
            float4 v = *(const float4*)(pc + i * 4);
            d[i * 4 + 0] = (_Float16)v.x; d[i * 4 + 1] = (_Float16)v.y;
            d[i * 4 + 2] = (_Float16)v.z; d[i * 4 + 3] = (_Float16)v.w;
          }
        }
        __builtin_prefetch(p + 64, 0, 3);
      }
    }
    __syncthreads();
    v16h a0 = frag_a16(sA[wm], lane),       b0 = frag_b16(sB[wn], lane);
    v16h a1 = frag_a16(sA[wm] + 512, lane), b1 = frag_b16(sB[wn] + 512, lane);
    acc = __builtin_amdgcn_wmma_f32_16x16x32_f16(false, a0, false, b0, (short)0, acc, false, false);
    acc = __builtin_amdgcn_wmma_f32_16x16x32_f16(false, a1, false, b1, (short)0, acc, false, false);
    __syncthreads();
  }
  // ---- 32-deep tail (e.g. K==32 attention scores) ----
  for (; k0 < g.K; k0 += 32) {
    if (wid < 2) {
      int row = lane & 15, kq = (lane >> 4) * 16;
      const float* p = A + (long)(mb + wid * 16 + row) * g.lda + k0 + kq;
      _Float16* d = &sA[wid][row * 32 + kq];
#pragma unroll
      for (int i = 0; i < 4; i++) {
        float4 v = *(const float4*)(p + i * 4);
        d[i * 4 + 0] = (_Float16)v.x; d[i * 4 + 1] = (_Float16)v.y;
        d[i * 4 + 2] = (_Float16)v.z; d[i * 4 + 3] = (_Float16)v.w;
      }
    } else {
      int t = wid - 2;
      if (!g.transB) {
        const float* p = B + (long)(k0 + lane) * g.ldb + nb + t * 16;
#pragma unroll
        for (int i = 0; i < 4; i++) {
          float4 v = *(const float4*)(p + i * 4);
          sB[t][(i * 4 + 0) * 32 + lane] = (_Float16)v.x;
          sB[t][(i * 4 + 1) * 32 + lane] = (_Float16)v.y;
          sB[t][(i * 4 + 2) * 32 + lane] = (_Float16)v.z;
          sB[t][(i * 4 + 3) * 32 + lane] = (_Float16)v.w;
        }
      } else {
        int row = lane & 15, kq = (lane >> 4) * 16;
        const float* p = B + (long)(nb + t * 16 + row) * g.ldb + k0 + kq;
        _Float16* d = &sB[t][row * 32 + kq];
#pragma unroll
        for (int i = 0; i < 4; i++) {
          float4 v = *(const float4*)(p + i * 4);
          d[i * 4 + 0] = (_Float16)v.x; d[i * 4 + 1] = (_Float16)v.y;
          d[i * 4 + 2] = (_Float16)v.z; d[i * 4 + 3] = (_Float16)v.w;
        }
      }
    }
    __syncthreads();
    v16h a = frag_a16(sA[wm], lane);
    v16h b = frag_b16(sB[wn], lane);
    acc = __builtin_amdgcn_wmma_f32_16x16x32_f16(false, a, false, b, (short)0, acc, false, false);
    __syncthreads();
  }
  int rbase = (lane >> 4) * 8;
  float* Cp = C + (long)(mb + wm * 16 + rbase) * g.ldc + nb + wn * 16 + (lane & 15);
#pragma unroll
  for (int r = 0; r < 8; r++) {
    float v = g.alpha * acc[r];
    if (g.beta != 0.f) v += g.beta * Cp[(long)r * g.ldc];
    Cp[(long)r * g.ldc] = v;
  }
}

static void gemm_go(hipStream_t st, const float* A, const float* B, float* C,
                    int M, int N, int K, int lda, int ldb, int ldc, int transB,
                    float alpha, float beta, int batches, int dv,
                    long a1, long a2, long b1, long b2, long c1, long c2) {
  GemmArgs g; g.A = A; g.B = B; g.C = C; g.M = M; g.N = N; g.K = K;
  g.lda = lda; g.ldb = ldb; g.ldc = ldc; g.transB = transB;
  g.alpha = alpha; g.beta = beta; g.div = dv;
  g.a1 = a1; g.a2 = a2; g.b1 = b1; g.b2 = b2; g.c1 = c1; g.c2 = c2;
  dim3 gr(N / 32, M / 32, batches);
  hipLaunchKernelGGL(gemm_kernel, gr, dim3(128), 0, st, g);
}

// ---------------- small kernels ----------------
__global__ __launch_bounds__(256) void gap_kernel(const float* x, float* out) {
  int bc = blockIdx.x, tid = threadIdx.x;
  __shared__ float red[256];
  float s = 0.f;
  for (int i = tid; i < Hh * Ww; i += 256) s += x[(long)bc * Hh * Ww + i];
  red[tid] = s; __syncthreads();
  for (int st = 128; st > 0; st >>= 1) { if (tid < st) red[tid] += red[tid + st]; __syncthreads(); }
  if (tid == 0) out[bc] = red[0] / (float)(Hh * Ww);
}

__global__ __launch_bounds__(256) void router_kernel(const float* gv, const float* gi,
    const float* Wr, const float* br, const float* pb, float* intent) {
  int tid = threadIdx.x, b = tid >> 6, j = tid & 63;
  __shared__ float spf[4][64];
  __shared__ float sw[4][4];
  float s = br[j];
  for (int c = 0; c < 64; c++)
    s += gv[b * 64 + c] * Wr[c * 64 + j] + gi[b * 64 + c] * Wr[(64 + c) * 64 + j];
  spf[b][j] = s; __syncthreads();
  if (tid < 16) {
    int bb = tid >> 2, i = tid & 3; float t = 0.f;
    for (int jj = 0; jj < 64; jj++) t += spf[bb][jj] * pb[i * 64 + jj];
    sw[bb][i] = t;
  }
  __syncthreads();
  if (tid < 4) {
    float mx = fmaxf(fmaxf(sw[tid][0], sw[tid][1]), fmaxf(sw[tid][2], sw[tid][3]));
    float e0 = __expf(sw[tid][0] - mx), e1 = __expf(sw[tid][1] - mx);
    float e2 = __expf(sw[tid][2] - mx), e3 = __expf(sw[tid][3] - mx);
    float inv = 1.f / (e0 + e1 + e2 + e3);
    sw[tid][0] = e0 * inv; sw[tid][1] = e1 * inv; sw[tid][2] = e2 * inv; sw[tid][3] = e3 * inv;
  }
  __syncthreads();
  float t = 0.f;
  for (int i = 0; i < 4; i++) t += sw[b][i] * pb[i * 64 + j];
  intent[b * 64 + j] = t;
}

__global__ __launch_bounds__(256) void genf_kernel(float* Fre, float* Fim) {
  int i = blockIdx.x * 256 + threadIdx.x;      // 65536
  int u = i >> 8, w = i & 255;
  int t = (u * w) & 255;                        // exact periodic reduction
  float ang = -6.283185307179586f * (float)t * (1.f / 256.f);
  Fre[i] = cosf(ang); Fim[i] = sinf(ang);
}

__global__ __launch_bounds__(256) void ampphase_kernel(float* re, float* im) {
  long n = S1;
  for (long i = (long)blockIdx.x * 256 + threadIdx.x; i < n; i += (long)gridDim.x * 256) {
    float r = re[i], m = im[i];
    re[i] = sqrtf(r * r + m * m);
    im[i] = atan2f(m, r);
  }
}

__global__ __launch_bounds__(256) void gmake_kernel(const float* fa, const float* fp,
                                                    float* gre, float* gim) {
  long n = S1;
  for (long i = (long)blockIdx.x * 256 + threadIdx.x; i < n; i += (long)gridDim.x * 256) {
    float a = fa[i], p = fp[i];
    gre[i] = a * cosf(p); gim[i] = a * sinf(p);
  }
}

__global__ __launch_bounds__(256) void avgmap_kernel(const float* a, const float* b, float* o) {
  long n = S1;
  for (long i = (long)blockIdx.x * 256 + threadIdx.x; i < n; i += (long)gridDim.x * 256)
    o[i] = 0.5f * (a[i] + b[i]);
}

// ---------------- score net: gathered-A WMMA GEMM + fused MLP head ----------------
__global__ __launch_bounds__(256) void score_kernel(const float* vmap, const float* imap,
    const float* intent, const float* W1, const float* b1, const float* W2, const float* b2,
    float* scores) {
  int b = blockIdx.y, tile = blockIdx.x;
  int tid = threadIdx.x, lane = tid & 31, wid = tid >> 5;
  __shared__ __align__(16) _Float16 sA[512];
  __shared__ __align__(16) _Float16 sB[128 * 32];
  __shared__ float ssum[16];
  if (tid < 16) ssum[tid] = 0.f;
  v8f acc = {};
  for (int k0 = 0; k0 < SFEAT; k0 += 32) {
    if (tid < 128) {
      int row = tid >> 3, q = tid & 7;
      int f = k0 + q * 4;
      int token = tile * 16 + row, py = token >> 6, px = token & 63;
      float4 v;
      if (f < 1024) {
        int c = f >> 4, rem = f & 15, dy = rem >> 2;
        v = *(const float4*)&vmap[(((long)(b * 64 + c) * 256) + py * 4 + dy) * 256 + px * 4];
      } else if (f < 2048) {
        int ff = f - 1024, c = ff >> 4, rem = ff & 15, dy = rem >> 2;
        v = *(const float4*)&imap[(((long)(b * 64 + c) * 256) + py * 4 + dy) * 256 + px * 4];
      } else {
        float e[4];
#pragma unroll
        for (int jj = 0; jj < 4; jj++) {
          int g = f + jj - 2048;
          e[jj] = (g == 0) ? (float)py * (1.f / 63.f)
                : (g == 1) ? (float)px * (1.f / 63.f)
                : (g < 66) ? intent[b * 64 + g - 2] : 0.f;
        }
        v.x = e[0]; v.y = e[1]; v.z = e[2]; v.w = e[3];
      }
      _Float16* d = &sA[row * 32 + q * 4];
      d[0] = (_Float16)v.x; d[1] = (_Float16)v.y; d[2] = (_Float16)v.z; d[3] = (_Float16)v.w;
    }
    {
      int kk = tid >> 3, n0 = (tid & 7) * 16;
      int kg = k0 + kk;
#pragma unroll
      for (int i = 0; i < 16; i++) {
        float w = (kg < SFEAT) ? W1[(long)kg * 128 + n0 + i] : 0.f;
        sB[(n0 + i) * 32 + kk] = (_Float16)w;
      }
    }
    __syncthreads();
    v16h a = frag_a16(sA, lane);
    v16h bf = frag_b16(sB + (wid * 16) * 32, lane);
    acc = __builtin_amdgcn_wmma_f32_16x16x32_f16(false, a, false, bf, (short)0, acc, false, false);
    __syncthreads();
  }
  int col = wid * 16 + (lane & 15);
  float w2c = W2[col], bb = b1[col];
  int rbase = (lane >> 4) * 8;
#pragma unroll
  for (int r = 0; r < 8; r++) {
    float h = acc[r] + bb; h = h > 0.f ? h : 0.f;
    atomicAdd(&ssum[rbase + r], h * w2c);
  }
  __syncthreads();
  if (tid < 16) scores[b * NTOK + tile * 16 + tid] = ssum[tid] + b2[0];
}

// ---------------- exact top-k (set semantics) via radix bisection ----------------
__global__ __launch_bounds__(256) void topk_kernel(const float* scores, int* sel) {
  int b = blockIdx.x, tid = threadIdx.x;
  __shared__ unsigned keys[NTOK];
  __shared__ int scnt;
  for (int i = tid; i < NTOK; i += 256) {
    unsigned u = __float_as_uint(scores[b * NTOK + i]);
    u = (u & 0x80000000u) ? ~u : (u | 0x80000000u);
    keys[i] = u;
  }
  __syncthreads();
  unsigned t = 0;
  for (int bit = 31; bit >= 0; bit--) {
    if (tid == 0) scnt = 0;
    __syncthreads();
    unsigned cand = t | (1u << bit);
    int lc = 0;
    for (int i = tid; i < NTOK; i += 256) lc += (keys[i] >= cand) ? 1 : 0;
    atomicAdd(&scnt, lc);
    __syncthreads();
    if (scnt >= KSEL) t = cand;
    __syncthreads();
  }
  if (tid == 0) scnt = 0;
  __syncthreads();
  for (int i = tid; i < NTOK; i += 256)
    if (keys[i] > t) { int p = atomicAdd(&scnt, 1); sel[b * KSEL + p] = i; }
  __syncthreads();
  for (int i = tid; i < NTOK; i += 256)
    if (keys[i] == t) { int p = atomicAdd(&scnt, 1); if (p < KSEL) sel[b * KSEL + p] = i; }
}

// ---------------- token-gather WMMA GEMM (q/k/v projections) ----------------
__global__ __launch_bounds__(256) void tokgemm_kernel(const float* map, const int* sel,
                                                      const float* W, float* out) {
  int b = blockIdx.y, tile = blockIdx.x;
  int tid = threadIdx.x, lane = tid & 31, wid = tid >> 5;
  __shared__ __align__(16) _Float16 sA[512];
  __shared__ __align__(16) _Float16 sB[128 * 32];
  v8f acc = {};
  for (int k0 = 0; k0 < TOK; k0 += 32) {
    if (tid < 128) {
      int row = tid >> 3, q = tid & 7;
      int f = k0 + q * 4;
      int token = sel[b * KSEL + tile * 16 + row];
      int py = token >> 6, px = token & 63;
      int c = f >> 4, rem = f & 15, dy = rem >> 2;
      float4 v = *(const float4*)&map[(((long)(b * 64 + c) * 256) + py * 4 + dy) * 256 + px * 4];
      _Float16* d = &sA[row * 32 + q * 4];
      d[0] = (_Float16)v.x; d[1] = (_Float16)v.y; d[2] = (_Float16)v.z; d[3] = (_Float16)v.w;
    }
    {
      int kk = tid >> 3, n0 = (tid & 7) * 16;
#pragma unroll
      for (int i = 0; i < 16; i++)
        sB[(n0 + i) * 32 + kk] = (_Float16)W[(long)(k0 + kk) * 128 + n0 + i];
    }
    __syncthreads();
    v16h a = frag_a16(sA, lane);
    v16h bf = frag_b16(sB + (wid * 16) * 32, lane);
    acc = __builtin_amdgcn_wmma_f32_16x16x32_f16(false, a, false, bf, (short)0, acc, false, false);
    __syncthreads();
  }
  int col = wid * 16 + (lane & 15);
  int rbase = (lane >> 4) * 8;
#pragma unroll
  for (int r = 0; r < 8; r++) {
    int m = tile * 16 + rbase + r;
    out[((long)b * KSEL + m) * 128 + col] = acc[r];
  }
}

__global__ __launch_bounds__(256) void softmax_kernel(float* S) {
  long row = blockIdx.x;
  float* p = S + row * 1024;
  int tid = threadIdx.x;
  __shared__ float red[256];
  float v[4]; float mx = -3.4e38f;
#pragma unroll
  for (int j = 0; j < 4; j++) { v[j] = p[tid * 4 + j]; mx = fmaxf(mx, v[j]); }
  red[tid] = mx; __syncthreads();
  for (int s = 128; s > 0; s >>= 1) { if (tid < s) red[tid] = fmaxf(red[tid], red[tid + s]); __syncthreads(); }
  mx = red[0]; __syncthreads();
  float sum = 0.f;
#pragma unroll
  for (int j = 0; j < 4; j++) { v[j] = __expf(v[j] - mx); sum += v[j]; }
  red[tid] = sum; __syncthreads();
  for (int s = 128; s > 0; s >>= 1) { if (tid < s) red[tid] += red[tid + s]; __syncthreads(); }
  float inv = 1.f / red[0];
#pragma unroll
  for (int j = 0; j < 4; j++) p[tid * 4 + j] = v[j] * inv;
}

// scatter: outmap[selected token elems] = gather(vmap token) + fs0
__global__ __launch_bounds__(256) void scatter_kernel(float* outmap, const float* vmap,
                                                      const float* fs0, const int* sel) {
  int b = blockIdx.y, j = blockIdx.x;
  int tok = sel[b * KSEL + j];
  int py = tok >> 6, px = tok & 63;
  for (int e = threadIdx.x; e < 1024; e += 256) {
    int c = e >> 4, rem = e & 15, dy = rem >> 2, dx = rem & 3;
    long addr = (((long)(b * 64 + c) * 256) + py * 4 + dy) * 256 + px * 4 + dx;
    outmap[addr] = vmap[addr] + fs0[((long)b * KSEL + j) * 1024 + e];
  }
}

// ---------------- implicit-GEMM WMMA 3x3 conv ----------------
// weight repack: wf[((s*64)+oc)*32 + kk], s = tap*2 + icHalf, ic = icHalf*32 + kk
__global__ __launch_bounds__(256) void wpack_kernel(const float* w, _Float16* wf) {
  int idx = blockIdx.x * 256 + threadIdx.x;    // 64*576 = 36864
  if (idx >= 64 * 576) return;
  int oc = idx / 576, k = idx % 576;
  int s = k >> 5, kk = k & 31;
  int tap = s >> 1, ic = (s & 1) * 32 + kk;
  wf[((long)s * 64 + oc) * 32 + kk] = (_Float16)w[((long)oc * 64 + ic) * 9 + tap];
}

// block: one image row y, 64 pixels (x0..x0+63), all 64 oc.
// M = pixels (4 tiles of 16), N = oc (4 tiles of 16), K = 576 (18 steps of 32).
// Weight slabs (4KB/step, already fragment-ordered f16) are double-buffered into LDS
// with async global->LDS copies (ASYNCcnt), overlapped with the current step's WMMAs.
__global__ __launch_bounds__(256) void convw_kernel(const float* in, const _Float16* wf,
    const float* bias, float* out, int mode, const float* vis, const float* ir) {
  int b = blockIdx.z, y = blockIdx.y, x0 = blockIdx.x * 64;
  int tid = threadIdx.x, lane = tid & 31, wid = tid >> 5;
  // sIn[(row*66 + col)*64 + ic], rows y-1..y+1, cols x0-1..x0+64, f16
  __shared__ __align__(4)  _Float16 sIn[3 * 66 * 64];
  __shared__ __align__(16) _Float16 sW[2][2048];          // [buf][oc*32 + kk]
  for (int u = tid; u < 64 * 3 * 66; u += 256) {
    int ic = u / 198, rc = u % 198;            // rc = row*66+col (coalesced along col)
    int row = rc / 66, col = rc % 66;
    int gy = y + row - 1, gx = x0 + col - 1;
    float v = 0.f;
    if (gy >= 0 && gy < 256 && gx >= 0 && gx < 256)
      v = in[(((long)b * 64 + ic) * 256 + gy) * 256 + gx];
    sIn[rc * 64 + ic] = (_Float16)v;
  }
  // async weight staging: 256 lanes x 16B = one 4KB k-slab
  auto stageW = [&](int dbuf, int s) {
    const _Float16* src = wf + (long)s * 2048 + tid * 8;          // 8 f16 = 16B / lane
    unsigned dst = (unsigned)(unsigned long long)&sW[dbuf][tid * 8]; // LDS byte offset
    asm volatile("global_load_async_to_lds_b128 %0, %1, off"
                 :: "v"(dst), "v"(src) : "memory");
  };
  stageW(0, 0);
  asm volatile("s_wait_asynccnt 0x0" ::: "memory");
  __syncthreads();
  int mt = wid & 3, ng = wid >> 2;             // wave: pixel tile mt, oc tiles {2ng, 2ng+1}
  v8f acc0 = {}, acc1 = {};
  int m = lane & 15, hk = (lane >> 4) * 16;
  int nl = lane & 15, kb = (lane >> 4) * 16;
  for (int s = 0; s < 18; s++) {
    int cur = s & 1;
    if (s + 1 < 18) stageW(cur ^ 1, s + 1);    // overlap next slab with this step's WMMAs
    int tap = s >> 1, icH = (s & 1) * 32;
    int dy = tap / 3, dx = tap % 3;
    // A fragment: element (m, kk) = sIn[(dy*66 + mt*16 + m + dx)*64 + icH + kk]
    const _Float16* abase = &sIn[((dy * 66) + mt * 16 + m + dx) * 64 + icH];
    v16h a; unsigned* au = (unsigned*)&a;
#pragma unroll
    for (int j = 0; j < 8; j++) {
      int kk = hk + j * 2;
      au[j] = *(const unsigned*)(abase + kk);
    }
    // B fragments from LDS-staged weights
    const _Float16* wp0 = &sW[cur][((ng * 2 + 0) * 16 + nl) * 32 + kb];
    const _Float16* wp1 = wp0 + 16 * 32;
    v16h b0, b1; unsigned* b0u = (unsigned*)&b0; unsigned* b1u = (unsigned*)&b1;
#pragma unroll
    for (int j = 0; j < 8; j++) {
      b0u[j] = *(const unsigned*)(wp0 + 2 * j);
      b1u[j] = *(const unsigned*)(wp1 + 2 * j);
    }
    acc0 = __builtin_amdgcn_wmma_f32_16x16x32_f16(false, a, false, b0, (short)0, acc0, false, false);
    acc1 = __builtin_amdgcn_wmma_f32_16x16x32_f16(false, a, false, b1, (short)0, acc1, false, false);
    if (s + 1 < 18) {
      asm volatile("s_wait_asynccnt 0x0" ::: "memory");
      __syncthreads();
    }
  }
  int rbase = (lane >> 4) * 8;
#pragma unroll
  for (int nt = 0; nt < 2; nt++) {
    int oc = (ng * 2 + nt) * 16 + nl;
    float bia = bias[oc];
#pragma unroll
    for (int r = 0; r < 8; r++) {
      int px = x0 + mt * 16 + rbase + r;
      long o = (((long)b * 64 + oc) * 256 + y) * 256 + px;
      float val = (nt ? acc1[r] : acc0[r]) + bia;
      if (mode == 0) val = val > 0.f ? val : 0.f;
      else val += 0.5f * (vis[o] + ir[o]);
      out[o] = val;
    }
  }
}

// ---------------- host orchestration ----------------
extern "C" void kernel_launch(void* const* d_in, const int* in_sizes, int n_in,
                              void* d_out, int out_size, void* d_ws, size_t ws_size,
                              hipStream_t stream) {
  const float* vis = (const float*)d_in[0];
  const float* ir  = (const float*)d_in[1];
  const float* pb  = (const float*)d_in[2];
  const float* Wr  = (const float*)d_in[3];
  const float* br  = (const float*)d_in[4];
  const float* aW1 = (const float*)d_in[5];  const float* ab1 = (const float*)d_in[6];
  const float* aW2 = (const float*)d_in[7];  const float* ab2 = (const float*)d_in[8];
  const float* aWq = (const float*)d_in[9];  const float* aWk = (const float*)d_in[10];
  const float* aWv = (const float*)d_in[11]; const float* aWo = (const float*)d_in[12];
  const float* pW1 = (const float*)d_in[13]; const float* pb1 = (const float*)d_in[14];
  const float* pW2 = (const float*)d_in[15]; const float* pb2 = (const float*)d_in[16];
  const float* pWq = (const float*)d_in[17]; const float* pWk = (const float*)d_in[18];
  const float* pWv = (const float*)d_in[19]; const float* pWo = (const float*)d_in[20];
  const float* c1w = (const float*)d_in[21]; const float* c1b = (const float*)d_in[22];
  const float* c2w = (const float*)d_in[23]; const float* c2b = (const float*)d_in[24];
  float* out = (float*)d_out;

  auto buf = [&](int i) { return (float*)((char*)d_ws + (size_t)i * S1 * 4); };
  float* small0 = (float*)((char*)d_ws + (size_t)7 * S1 * 4);
  long off = 0;
  float* Fre    = small0 + off; off += 65536;
  float* Fim    = small0 + off; off += 65536;
  float* gv     = small0 + off; off += 256;
  float* gi     = small0 + off; off += 256;
  float* intent = small0 + off; off += 256;
  float* scores = small0 + off; off += Bb * NTOK;
  int*   sel    = (int*)(small0 + off); off += Bb * KSEL;
  float* qb     = small0 + off; off += (long)Bb * KSEL * 128;
  float* kb     = small0 + off; off += (long)Bb * KSEL * 128;
  float* vb     = small0 + off; off += (long)Bb * KSEL * 128;
  float* ob     = small0 + off; off += (long)Bb * KSEL * 128;
  float* fs0    = small0 + off; off += (long)Bb * KSEL * 1024;
  _Float16* wf1 = (_Float16*)(small0 + off); off += 18432;   // 36864 f16
  _Float16* wf2 = (_Float16*)(small0 + off); off += 18432;

  const long IM = (long)Hh * Ww;   // 65536 per (b,c) image

  // router + constants
  hipLaunchKernelGGL(gap_kernel, dim3(Bb * Cc), dim3(256), 0, stream, vis, gv);
  hipLaunchKernelGGL(gap_kernel, dim3(Bb * Cc), dim3(256), 0, stream, ir, gi);
  hipLaunchKernelGGL(router_kernel, dim3(1), dim3(256), 0, stream, gv, gi, Wr, br, pb, intent);
  hipLaunchKernelGGL(genf_kernel, dim3(256), dim3(256), 0, stream, Fre, Fim);
  hipLaunchKernelGGL(wpack_kernel, dim3(144), dim3(256), 0, stream, c1w, wf1);
  hipLaunchKernelGGL(wpack_kernel, dim3(144), dim3(256), 0, stream, c2w, wf2);

  // forward DFT: rows then columns (F symmetric)
  auto fwd = [&](const float* X, float* oRe, float* oIm) {
    float* Yre = buf(4); float* Yim = buf(5);
    gemm_go(stream, X, Fre, Yre, Bb * Cc * Hh, 256, 256, 256, 256, 256, 0, 1.f, 0.f, 1, 1, 0,0,0,0,0,0);
    gemm_go(stream, X, Fim, Yim, Bb * Cc * Hh, 256, 256, 256, 256, 256, 0, 1.f, 0.f, 1, 1, 0,0,0,0,0,0);
    gemm_go(stream, Fre, Yre, oRe, 256, 256, 256, 256, 256, 256, 0,  1.f, 0.f, Bb * Cc, 1, 0,0, IM,0, IM,0);
    gemm_go(stream, Fim, Yim, oRe, 256, 256, 256, 256, 256, 256, 0, -1.f, 1.f, Bb * Cc, 1, 0,0, IM,0, IM,0);
    gemm_go(stream, Fre, Yim, oIm, 256, 256, 256, 256, 256, 256, 0,  1.f, 0.f, Bb * Cc, 1, 0,0, IM,0, IM,0);
    gemm_go(stream, Fim, Yre, oIm, 256, 256, 256, 256, 256, 256, 0,  1.f, 1.f, Bb * Cc, 1, 0,0, IM,0, IM,0);
  };
  fwd(vis, buf(0), buf(1));
  fwd(ir,  buf(2), buf(3));
  hipLaunchKernelGGL(ampphase_kernel, dim3(8192), dim3(256), 0, stream, buf(0), buf(1)); // va, vp
  hipLaunchKernelGGL(ampphase_kernel, dim3(8192), dim3(256), 0, stream, buf(2), buf(3)); // ia, ip

  // fuse branch
  auto branch = [&](const float* vm, const float* im_,
                    const float* W1, const float* b1, const float* W2, const float* b2,
                    const float* Wq, const float* Wk, const float* Wv, const float* Wo,
                    float* outMap) {
    hipLaunchKernelGGL(score_kernel, dim3(NTOK / 16, Bb), dim3(256), 0, stream,
                       vm, im_, intent, W1, b1, W2, b2, scores);
    hipLaunchKernelGGL(topk_kernel, dim3(Bb), dim3(256), 0, stream, scores, sel);
    hipLaunchKernelGGL(tokgemm_kernel, dim3(KSEL / 16, Bb), dim3(256), 0, stream, vm,  sel, Wq, qb);
    hipLaunchKernelGGL(tokgemm_kernel, dim3(KSEL / 16, Bb), dim3(256), 0, stream, im_, sel, Wk, kb);
    hipLaunchKernelGGL(tokgemm_kernel, dim3(KSEL / 16, Bb), dim3(256), 0, stream, im_, sel, Wv, vb);
    // S = (q @ k^T) / sqrt(dh) : per (b,h), K = 32 (tail path)
    float* Sb = buf(6);
    gemm_go(stream, qb, kb, Sb, 1024, 1024, 32, 128, 128, 1024, 1, 0.1767766953f, 0.f,
            Bb * NHh, NHh, 131072, 32, 131072, 32, 4194304, 1048576);
    hipLaunchKernelGGL(softmax_kernel, dim3(Bb * NHh * 1024), dim3(256), 0, stream, Sb);
    // o = P @ v : per (b,h)
    gemm_go(stream, Sb, vb, ob, 1024, 32, 1024, 1024, 128, 128, 0, 1.f, 0.f,
            Bb * NHh, NHh, 4194304, 1048576, 131072, 32, 131072, 32);
    // fs0 = o @ Wo : per b
    gemm_go(stream, ob, Wo, fs0, 1024, 1024, 128, 128, 1024, 1024, 0, 1.f, 0.f,
            Bb, 1, 131072, 0, 0, 0, 1048576, 0);
    hipLaunchKernelGGL(avgmap_kernel, dim3(8192), dim3(256), 0, stream, vm, im_, outMap);
    hipLaunchKernelGGL(scatter_kernel, dim3(KSEL, Bb), dim3(256), 0, stream, outMap, vm, fs0, sel);
  };
  branch(buf(0), buf(2), aW1, ab1, aW2, ab2, aWq, aWk, aWv, aWo, buf(4)); // fa
  branch(buf(1), buf(3), pW1, pb1, pW2, pb2, pWq, pWk, pWv, pWo, buf(5)); // fp

  // inverse DFT of G = fa * exp(i*fp); only real part needed.
  hipLaunchKernelGGL(gmake_kernel, dim3(8192), dim3(256), 0, stream, buf(4), buf(5), buf(0), buf(1));
  const float s = 1.f / 256.f;
  // columns: H1 = Finv @ G ; Finv = conj(F)/256
  gemm_go(stream, Fre, buf(0), buf(2), 256, 256, 256, 256, 256, 256, 0,  s, 0.f, Bb * Cc, 1, 0,0, IM,0, IM,0);
  gemm_go(stream, Fim, buf(1), buf(2), 256, 256, 256, 256, 256, 256, 0,  s, 1.f, Bb * Cc, 1, 0,0, IM,0, IM,0);
  gemm_go(stream, Fre, buf(1), buf(3), 256, 256, 256, 256, 256, 256, 0,  s, 0.f, Bb * Cc, 1, 0,0, IM,0, IM,0);
  gemm_go(stream, Fim, buf(0), buf(3), 256, 256, 256, 256, 256, 256, 0, -s, 1.f, Bb * Cc, 1, 0,0, IM,0, IM,0);
  // rows: spatial = Re(H1 @ Finv) = (H1re@Fre + H1im@Fim)/256
  gemm_go(stream, buf(2), Fre, buf(4), Bb * Cc * Hh, 256, 256, 256, 256, 256, 0, s, 0.f, 1, 1, 0,0,0,0,0,0);
  gemm_go(stream, buf(3), Fim, buf(4), Bb * Cc * Hh, 256, 256, 256, 256, 256, 0, s, 1.f, 1, 1, 0,0,0,0,0,0);

  // conv stack + residual (implicit-GEMM WMMA, async weight staging)
  hipLaunchKernelGGL(convw_kernel, dim3(4, 256, Bb), dim3(256), 0, stream,
                     buf(4), wf1, c1b, buf(5), 0, nullptr, nullptr);
  hipLaunchKernelGGL(convw_kernel, dim3(4, 256, Bb), dim3(256), 0, stream,
                     buf(5), wf2, c2b, out, 1, vis, ir);
}
